// LSTMTagger_48361331753727
// MI455X (gfx1250) — matrix operand order repaired and checked by reading, not compiled, not run on hardware
//
#include <hip/hip_runtime.h>
#include <math.h>

// ---------------- problem constants (must match reference) ----------------
#define S_LEN 8192
#define L_CH  16
#define CDIM  256
#define WDIM  512
#define HID   1024
#define TAGS  128
#define XDIM  (WDIM + CDIM)   // 768

typedef __bf16 bf16_t;
typedef bf16_t bf16x8  __attribute__((ext_vector_type(8)));
typedef bf16_t bf16x16 __attribute__((ext_vector_type(16)));
typedef float  f32x8   __attribute__((ext_vector_type(8)));

__device__ __forceinline__ f32x8 splat8(float x) {
  f32x8 v;
#pragma unroll
  for (int i = 0; i < 8; ++i) v[i] = x;
  return v;
}

__device__ __forceinline__ bf16x16 cat16(bf16x8 lo, bf16x8 hi) {
  return __builtin_shufflevector(lo, hi, 0,1,2,3,4,5,6,7,8,9,10,11,12,13,14,15);
}

// A fragment, 16x32 bf16 (MxK). Source row-major, K contiguous.
// lane: m = lane&15, hf = lane>>4.  VGPR0..3 = K [hf*8, hf*8+8),
// VGPR4..7 = K [16+hf*8, 16+hf*8+8)  -> two 16B loads.
__device__ __forceinline__ bf16x16 frag_a(const bf16_t* row, int k0, int hf) {
  const bf16_t* p = row + k0 + hf * 8;
  return cat16(*(const bf16x8*)p, *(const bf16x8*)(p + 16));
}
// B fragment, 32x16 bf16 (KxN) taken from row-major W[n][k] (z = x * W^T).
// lane: n = lane&15, hf = lane>>4.  K [hf*8, hf*8+8) then [16+hf*8, 16+hf*8+8).
__device__ __forceinline__ bf16x16 frag_b(const bf16_t* row, int k0, int hf) {
  const bf16_t* p = row + k0 + hf * 8;
  return cat16(*(const bf16x8*)p, *(const bf16x8*)(p + 16));
}

__device__ __forceinline__ f32x8 wmma_bf16(bf16x16 a, bf16x16 b, f32x8 c) {
  // 8 args: (neg_a, A, neg_b, B, c_mod, C, reuse_a, reuse_b)
  return __builtin_amdgcn_wmma_f32_16x16x32_bf16(false, a, false, b,
                                                 (short)0, c, false, false);
}

__device__ __forceinline__ float sigm(float x) { return 1.f / (1.f + __expf(-x)); }

// ---------------- small utility kernels ----------------
__global__ void k_cvt_bf16(const float* __restrict__ in, bf16_t* __restrict__ out, int n) {
  int i = blockIdx.x * blockDim.x + threadIdx.x;
  if (i < n) out[i] = (bf16_t)in[i];
}

__global__ void k_zero32(unsigned* __restrict__ p, int n) {
  int i = blockIdx.x * blockDim.x + threadIdx.x;
  if (i < n) p[i] = 0u;
}

// Xw[s][0:512] = bf16(word_emb[word_idxs[s]])
__global__ __launch_bounds__(WDIM) void k_gather_words(
    const int* __restrict__ word_idxs, const float* __restrict__ word_emb,
    bf16_t* __restrict__ Xw) {
  int s = blockIdx.x, c = threadIdx.x;
  int wi = word_idxs[s];
  Xw[(size_t)s * XDIM + c] = (bf16_t)word_emb[(size_t)wi * WDIM + c];
}

// ---------------- char-level LSTM (WMMA, fused over 16 time steps) -------
// Block = 1024 threads = 32 waves; block owns 16 words (rows) and the full
// 256-wide hidden state.  Wave w: column tile jt = w&15 (16 hidden units),
// quarter pair qp = w>>4 (0 -> gates i,f ; 1 -> gates g,o).  Two f32x8
// accumulators per wave; the (g,o) waves hand their z tiles to the (i,f)
// waves via LDS for the gate update; c state lives in (i,f) registers.
// The K-loop is FORCED rolled (#pragma unroll 1): the per-wave weight
// working set (32KB = 256 VGPRs) can never be register-resident, so the
// correct behavior is to re-stream it from the 192MB L2 each K-step; a
// rolled loop stops unroll+LICM from creating the 32-fragment live set
// that caused scratch spills in the previous two rounds.
__global__ __launch_bounds__(1024) void k_char_lstm(
    const int* __restrict__ char_idxs, const int* __restrict__ char_lens,
    const float* __restrict__ char_emb, const bf16_t* __restrict__ Wih,
    const bf16_t* __restrict__ Whh, const float* __restrict__ b_c,
    bf16_t* __restrict__ Xw) {
  __shared__ __align__(16) bf16_t Xbuf[16][CDIM];
  __shared__ __align__(16) bf16_t Hbuf[16][CDIM];
  __shared__ __align__(16) float  Zbuf[2][16][CDIM];   // z_g, z_o tiles

  const int tid  = threadIdx.x;
  const int lane = tid & 31, w = tid >> 5;
  const int jt = w & 15;              // hidden column tile
  const int qp = w >> 4;              // 0: i,f   1: g,o
  const int hf = lane >> 4, r16 = lane & 15;
  const int s0 = blockIdx.x * 16;
  const int srow = tid >> 6;          // staging row (0..15)
  const int scol = (tid & 63) * 4;    // staging col (4 elems / thread)

  for (int i = tid; i < 16 * CDIM; i += 1024) ((bf16_t*)Hbuf)[i] = (bf16_t)0.f;

  const int q0 = qp * 2, q1 = qp * 2 + 1;
  const float bias0 = b_c[q0 * CDIM + jt * 16 + r16];
  const float bias1 = b_c[q1 * CDIM + jt * 16 + r16];
  const bf16_t* brow0_ih = Wih + (size_t)(q0 * CDIM + jt * 16 + r16) * CDIM;
  const bf16_t* brow1_ih = Wih + (size_t)(q1 * CDIM + jt * 16 + r16) * CDIM;
  const bf16_t* brow0_hh = Whh + (size_t)(q0 * CDIM + jt * 16 + r16) * CDIM;
  const bf16_t* brow1_hh = Whh + (size_t)(q1 * CDIM + jt * 16 + r16) * CDIM;
  f32x8 cst = {};
  const int mylen = char_lens[s0 + srow] - 1;

#pragma unroll 1
  for (int t = 0; t < L_CH; ++t) {
    __syncthreads();
    {  // stage x_t = char_emb[char_idxs[s][t]] as bf16
      int ci = char_idxs[(s0 + srow) * L_CH + t];
      const float* src = char_emb + (size_t)ci * CDIM + scol;
#pragma unroll
      for (int e = 0; e < 4; ++e) Xbuf[srow][scol + e] = (bf16_t)src[e];
    }
    __syncthreads();

    f32x8 acc0 = splat8(bias0);
    f32x8 acc1 = splat8(bias1);

#pragma unroll 1
    for (int kk = 0; kk < CDIM / 32; ++kk) {
      const int k0 = kk * 32;
      bf16x16 ax = frag_a(&Xbuf[r16][0], k0, hf);
      bf16x16 ah = frag_a(&Hbuf[r16][0], k0, hf);
      // two independent accumulator chains interleave -> hazard NOPs filled
      acc0 = wmma_bf16(ax, frag_b(brow0_ih, k0, hf), acc0);
      acc1 = wmma_bf16(ax, frag_b(brow1_ih, k0, hf), acc1);
      acc0 = wmma_bf16(ah, frag_b(brow0_hh, k0, hf), acc0);
      acc1 = wmma_bf16(ah, frag_b(brow1_hh, k0, hf), acc1);
      // no software pipelining across K-steps (keeps <=4 B-frags live)
      __builtin_amdgcn_sched_barrier(0);
    }
    __syncthreads();
    if (qp == 1) {  // publish z_g, z_o tiles
#pragma unroll
      for (int r = 0; r < 8; ++r) {
        Zbuf[0][hf * 8 + r][jt * 16 + r16] = acc0[r];
        Zbuf[1][hf * 8 + r][jt * 16 + r16] = acc1[r];
      }
    }
    __syncthreads();
    if (qp == 0) {  // full gate update for 16m x 16j tile
#pragma unroll
      for (int r = 0; r < 8; ++r) {
        const int m = hf * 8 + r, j = jt * 16 + r16;
        float g = Zbuf[0][m][j], o = Zbuf[1][m][j];
        float c = sigm(acc1[r]) * cst[r] + sigm(acc0[r]) * tanhf(g);
        cst[r] = c;
        Hbuf[m][j] = (bf16_t)(sigm(o) * tanhf(c));
      }
    }
    __syncthreads();
    if (mylen == t) {  // emit char_feat into word-LSTM input cols 512..767
      bf16_t* dst = Xw + (size_t)(s0 + srow) * XDIM + WDIM + scol;
#pragma unroll
      for (int e = 0; e < 4; ++e) dst[e] = Hbuf[srow][scol + e];
    }
  }
}

// ---------------- word-level LSTM (serial, 64 WGs, LDS-resident weights) --
#define NWG  64
#define ROWS 64  // gate rows per WG = 4 gates x 16 hidden units
__global__ __launch_bounds__(256) void k_word_lstm(
    const bf16_t* __restrict__ Wih, const bf16_t* __restrict__ Whh,
    const float* __restrict__ b_w, const bf16_t* __restrict__ Xw,
    bf16_t* __restrict__ Hh, unsigned* __restrict__ counters) {
  extern __shared__ char smem[];
  const int WROW = XDIM + HID;                         // 1792
  bf16_t* Wl      = (bf16_t*)smem;                     // [ROWS][1792] = 224KB
  float*  partial = (float*)(smem + (size_t)ROWS * WROW * sizeof(bf16_t));
  float*  zbuf    = partial + 256;
  float*  cbuf    = zbuf + ROWS;

  const int tid = threadIdx.x;
  const int j0  = blockIdx.x * 16;  // this WG's hidden-unit slice

  // Stage the bf16 weight slice once; it stays in LDS for all 8192 steps
  // (only possible because CDNA5 has 320KB LDS per WGP).
  for (int idx = tid; idx < ROWS * WROW; idx += 256) {
    int lr = idx / WROW, col = idx - lr * WROW;
    int grow = (lr >> 4) * HID + j0 + (lr & 15);       // gate q, unit j
    Wl[idx] = (col < XDIM) ? Wih[(size_t)grow * XDIM + col]
                           : Whh[(size_t)grow * HID + (col - XDIM)];
  }
  if (tid < 16) cbuf[tid] = 0.f;
  __syncthreads();

  const int lr = tid >> 2, seg = tid & 3;  // 4 threads per gate row
  const int SEG = WROW / 4;                // 448 columns each
  const int c0 = seg * SEG;
  const bf16_t* wrow = Wl + (size_t)lr * WROW;
  const float bw = (tid < ROWS) ? b_w[(tid >> 4) * HID + j0 + (tid & 15)] : 0.f;

#pragma unroll 1
  for (int t = 0; t < S_LEN; ++t) {
    const bf16_t* xrow = Xw + (size_t)t * XDIM;
    const bf16_t* hrow = Hh + (size_t)(t - 1) * HID;  // valid when t>0
    float acc = 0.f;
    int xend = c0 + SEG; if (xend > XDIM) xend = XDIM;
    for (int col = c0; col < xend; col += 8) {
      bf16x8 w8 = *(const bf16x8*)(wrow + col);
      bf16x8 x8 = *(const bf16x8*)(xrow + col);
#pragma unroll
      for (int e = 0; e < 8; ++e) acc += (float)w8[e] * (float)x8[e];
    }
    if (t > 0) {
      int hs = c0; if (hs < XDIM) hs = XDIM;
      for (int col = hs; col < c0 + SEG; col += 8) {
        bf16x8 w8 = *(const bf16x8*)(wrow + col);
        bf16x8 h8 = *(const bf16x8*)(hrow + (col - XDIM));
#pragma unroll
        for (int e = 0; e < 8; ++e) acc += (float)w8[e] * (float)h8[e];
      }
    }
    partial[tid] = acc;
    __syncthreads();
    if (tid < ROWS)
      zbuf[tid] = partial[tid * 4] + partial[tid * 4 + 1] +
                  partial[tid * 4 + 2] + partial[tid * 4 + 3] + bw;
    __syncthreads();
    if (tid < 16) {
      float c = sigm(zbuf[16 + tid]) * cbuf[tid] +
                sigm(zbuf[tid]) * tanhf(zbuf[32 + tid]);
      cbuf[tid] = c;
      Hh[(size_t)t * HID + j0 + tid] = (bf16_t)(sigm(zbuf[48 + tid]) * tanhf(c));
    }
    __threadfence();     // make h slice visible at device scope
    __syncthreads();     // order tid0's arrive after the stores above
    if (tid == 0) {
      __hip_atomic_fetch_add(&counters[t], 1u, __ATOMIC_ACQ_REL,
                             __HIP_MEMORY_SCOPE_AGENT);
      while (__hip_atomic_load(&counters[t], __ATOMIC_ACQUIRE,
                               __HIP_MEMORY_SCOPE_AGENT) < NWG)
        __builtin_amdgcn_s_sleep(2);
    }
    __syncthreads();
  }
}

// ---------------- tag projection: logits = Hh * W_tag^T + b (WMMA) -------
__global__ __launch_bounds__(256) void k_logits(
    const bf16_t* __restrict__ Hh, const bf16_t* __restrict__ Wt,
    const float* __restrict__ b_tag, float* __restrict__ logits) {
  const int tid = threadIdx.x;
  const int lane = tid & 31, wv = tid >> 5;  // 8 waves -> 8 n-tiles = 128 tags
  const int hf = lane >> 4, r16 = lane & 15;
  const int m0 = blockIdx.x * 16;
  const int n = wv * 16 + r16;
  // two independent accumulator chains (even/odd K-steps) to fill NOPs
  f32x8 accA = splat8(b_tag[n]);
  f32x8 accB = {};
  const bf16_t* arow = Hh + (size_t)(m0 + r16) * HID;
  const bf16_t* brow = Wt + (size_t)n * HID;
#pragma unroll 1
  for (int kk = 0; kk < HID / 32; kk += 2) {
    accA = wmma_bf16(frag_a(arow, kk * 32, hf),
                     frag_b(brow, kk * 32, hf), accA);
    accB = wmma_bf16(frag_a(arow, (kk + 1) * 32, hf),
                     frag_b(brow, (kk + 1) * 32, hf), accB);
    __builtin_amdgcn_sched_barrier(0);
  }
#pragma unroll
  for (int r = 0; r < 8; ++r)
    logits[(size_t)(m0 + hf * 8 + r) * TAGS + n] = accA[r] + accB[r];
}

// ---------------- row-wise log-softmax over 128 tags ----------------------
__global__ __launch_bounds__(TAGS) void k_logsoftmax(
    const float* __restrict__ logits, float* __restrict__ out) {
  __shared__ float red[TAGS];
  int rI = blockIdx.x, t = threadIdx.x;
  float v = logits[(size_t)rI * TAGS + t];
  red[t] = v;
  __syncthreads();
  for (int s = TAGS / 2; s > 0; s >>= 1) {
    if (t < s) red[t] = fmaxf(red[t], red[t + s]);
    __syncthreads();
  }
  float mx = red[0];
  __syncthreads();
  red[t] = __expf(v - mx);
  __syncthreads();
  for (int s = TAGS / 2; s > 0; s >>= 1) {
    if (t < s) red[t] += red[t + s];
    __syncthreads();
  }
  float lse = __logf(red[0]);
  out[(size_t)rI * TAGS + t] = v - mx - lse;
}

// ---------------- host-side orchestration --------------------------------
extern "C" void kernel_launch(void* const* d_in, const int* in_sizes, int n_in,
                              void* d_out, int out_size, void* d_ws, size_t ws_size,
                              hipStream_t stream) {
  (void)in_sizes; (void)n_in; (void)out_size; (void)ws_size;
  const int*   word_idxs = (const int*)d_in[0];
  const int*   char_idxs = (const int*)d_in[1];
  const int*   char_lens = (const int*)d_in[2];
  const float* char_emb  = (const float*)d_in[3];
  const float* Wih_c     = (const float*)d_in[4];
  const float* Whh_c     = (const float*)d_in[5];
  const float* b_c       = (const float*)d_in[6];
  const float* word_emb  = (const float*)d_in[7];
  const float* Wih_w     = (const float*)d_in[8];
  const float* Whh_w     = (const float*)d_in[9];
  const float* b_w       = (const float*)d_in[10];
  const float* W_tag     = (const float*)d_in[11];
  const float* b_tag     = (const float*)d_in[12];

  char* ws = (char*)d_ws;
  size_t off = 0;
  auto carve = [&](size_t bytes) -> char* {
    char* p = ws + off;
    off += (bytes + 255) & ~(size_t)255;
    return p;
  };
  const int nWcih = 4 * CDIM * CDIM;        // 262144
  const int nWchh = 4 * CDIM * CDIM;        // 262144
  const int nWwih = 4 * HID * XDIM;         // 3145728
  const int nWwhh = 4 * HID * HID;          // 4194304
  const int nWtag = TAGS * HID;             // 131072
  bf16_t*   Wc_ih = (bf16_t*)carve((size_t)nWcih * sizeof(bf16_t));
  bf16_t*   Wc_hh = (bf16_t*)carve((size_t)nWchh * sizeof(bf16_t));
  bf16_t*   Ww_ih = (bf16_t*)carve((size_t)nWwih * sizeof(bf16_t));
  bf16_t*   Ww_hh = (bf16_t*)carve((size_t)nWwhh * sizeof(bf16_t));
  bf16_t*   Wt    = (bf16_t*)carve((size_t)nWtag * sizeof(bf16_t));
  bf16_t*   Xw    = (bf16_t*)carve((size_t)S_LEN * XDIM * sizeof(bf16_t));
  bf16_t*   Hh    = (bf16_t*)carve((size_t)S_LEN * HID * sizeof(bf16_t));
  unsigned* cnt   = (unsigned*)carve((size_t)S_LEN * sizeof(unsigned));
  float*    lg    = (float*)carve((size_t)S_LEN * TAGS * sizeof(float));

  // 1) fp32 -> bf16 weight copies
  k_cvt_bf16<<<(nWcih + 255) / 256, 256, 0, stream>>>(Wih_c, Wc_ih, nWcih);
  k_cvt_bf16<<<(nWchh + 255) / 256, 256, 0, stream>>>(Whh_c, Wc_hh, nWchh);
  k_cvt_bf16<<<(nWwih + 255) / 256, 256, 0, stream>>>(Wih_w, Ww_ih, nWwih);
  k_cvt_bf16<<<(nWwhh + 255) / 256, 256, 0, stream>>>(Whh_w, Ww_hh, nWwhh);
  k_cvt_bf16<<<(nWtag + 255) / 256, 256, 0, stream>>>(W_tag, Wt, nWtag);
  // 2) per-launch counter reset (deterministic / capture-safe)
  k_zero32<<<(S_LEN + 255) / 256, 256, 0, stream>>>(cnt, S_LEN);
  // 3) word embedding gather into Xw[:, 0:512]
  k_gather_words<<<S_LEN, WDIM, 0, stream>>>(word_idxs, word_emb, Xw);
  // 4) char LSTM (WMMA) -> Xw[:, 512:768]
  k_char_lstm<<<S_LEN / 16, 1024, 0, stream>>>(char_idxs, char_lens, char_emb,
                                               Wc_ih, Wc_hh, b_c, Xw);
  // 5) serial word LSTM, 64 cooperating WGs, LDS-resident weights
  size_t smem = (size_t)ROWS * (XDIM + HID) * sizeof(bf16_t) +
                (256 + ROWS + 16) * sizeof(float);
  k_word_lstm<<<NWG, 256, smem, stream>>>(Ww_ih, Ww_hh, b_w, Xw, Hh, cnt);
  // 6) tag GEMM (WMMA) + 7) log-softmax
  k_logits<<<S_LEN / 16, 256, 0, stream>>>(Hh, Wt, b_tag, lg);
  k_logsoftmax<<<S_LEN, TAGS, 0, stream>>>(lg, (float*)d_out);
}
